// TemporalEncoderLayer_11905649344767
// MI455X (gfx1250) — matrix operand order repaired and checked by reading, not compiled, not run on hardware
//
#include <hip/hip_runtime.h>
#include <math.h>

// ---------------------------------------------------------------------------
// Types for CDNA5 WMMA (wave32)
// ---------------------------------------------------------------------------
typedef __attribute__((ext_vector_type(16))) __bf16 v16bf;
typedef __attribute__((ext_vector_type(8)))  __bf16 v8bf;
typedef __attribute__((ext_vector_type(8)))  float  v8f;
typedef __attribute__((address_space(3)))    __bf16 lds_bf16;

#define HH 16
#define DD 32
#define SS 16
#define LL 1024
#define NN 4
#define CC 512
#define TOKENS (NN * LL)          // 4096
#define EPS_PN 1e-5f
#define TEMP_SCALE 0.42044820762685725f   // 32^(-1/4)

// GEMM tiling
#define BM 256                    // block rows   (8 waves x 32)
#define BN 64                     // block cols
#define BK 32                     // k-step (WMMA K)
#define LDS_STRIDE 40             // bf16 elems/row: 64B payload + 16B pad (bank-conflict-free)
#define A_ELEMS (BM * LDS_STRIDE) // 10240 bf16
#define B_ELEMS (BN * LDS_STRIDE) // 2560  bf16
#define BUF_ELEMS (A_ELEMS + B_ELEMS)       // 12800 bf16 / buffer
#define A_BYTES (A_ELEMS * 2)
#define BUF_BYTES (BUF_ELEMS * 2)           // 25600 B; x2 buffers = 50 KiB LDS

// ---------------------------------------------------------------------------
// fp32 -> bf16 conversion
// ---------------------------------------------------------------------------
__global__ void f32_to_bf16_kernel(const float* __restrict__ in,
                                   __bf16* __restrict__ out, int n) {
    int i = blockIdx.x * blockDim.x + threadIdx.x;
    if (i < n) out[i] = (__bf16)in[i];
}

// ---------------------------------------------------------------------------
// Segment boundaries from sorted bi (values in [0,16))
// ---------------------------------------------------------------------------
__global__ void seg_bounds_kernel(const int* __restrict__ bi,
                                  int* __restrict__ segoff) {
    int s = threadIdx.x;
    if (s <= SS) {
        int cnt = 0;
        for (int i = 0; i < LL; ++i) cnt += (bi[i] < s) ? 1 : 0;
        segoff[s] = cnt;
    }
}

// ---------------------------------------------------------------------------
// CDNA5 async global->LDS copy (16B per lane), ASYNCcnt-tracked
// ---------------------------------------------------------------------------
__device__ inline void async_g2l_b128(unsigned lds_off, const __bf16* gsrc) {
    asm volatile("global_load_async_to_lds_b128 %0, %1, off"
                 :: "v"(lds_off), "v"(gsrc)
                 : "memory");
}
__device__ inline void wait_async0() {
    asm volatile("s_wait_asynccnt 0" ::: "memory");
}

// ---------------------------------------------------------------------------
// LDS fragment loaders (16-bit A 16x32 and B 32x16 layouts, ISA 7.12.2)
// ---------------------------------------------------------------------------
__device__ inline v16bf lds_a_frag(const __bf16* __restrict__ sA, int row0, int lane) {
    int row = row0 + (lane & 15);
    int kb  = (lane >> 4) * 8;
    const __bf16* p = sA + row * LDS_STRIDE + kb;
    v8bf lo = *(const v8bf*)(p);
    v8bf hi = *(const v8bf*)(p + 16);
    v16bf a;
#pragma unroll
    for (int i = 0; i < 8; ++i) { a[i] = lo[i]; a[8 + i] = hi[i]; }
    return a;
}

__device__ inline v16bf lds_b_frag(const __bf16* __restrict__ sB, int n0, int lane) {
    int row = n0 + (lane & 15);
    int kb  = (lane >> 4) * 8;
    const __bf16* p = sB + row * LDS_STRIDE + kb;
    v8bf lo = *(const v8bf*)(p);
    v8bf hi = *(const v8bf*)(p + 16);
    v16bf b;
#pragma unroll
    for (int i = 0; i < 8; ++i) { b[i] = lo[i]; b[8 + i] = hi[i]; }
    return b;
}

// ---------------------------------------------------------------------------
// GEMM: Y = A(M x K, bf16 row-major) * W(N x K, bf16 row-major)^T  [+ bias]
// 256x64 block tile, 8 waves; wave = 32 rows x 64 cols -> 8 WMMA / k-step.
// A/B tiles double-buffered in LDS via async global->LDS copies.
// MODE 0: phi() on first 1024 columns, fp32 out (QKV)
// MODE 1: +bias, exact GELU, bf16 out (FFN-1)
// MODE 2: +bias, fp32 out (FFN-2)
// ---------------------------------------------------------------------------
template <int MODE>
__global__ void gemm_bf16_wmma_kernel(const __bf16* __restrict__ A,
                                      const __bf16* __restrict__ B,
                                      const float* __restrict__ bias,
                                      float* __restrict__ Cf,
                                      __bf16* __restrict__ Cb,
                                      int M, int N, int K) {
    __shared__ __attribute__((aligned(16))) __bf16 smem[2 * BUF_ELEMS];

    const int tid  = threadIdx.x;
    const int lane = tid & 31;
    const int wave = tid >> 5;                 // 0..7
    const int m0   = blockIdx.x * BM;
    const int n0   = blockIdx.y * BN;
    const int mw   = wave * 32;                // wave's row offset in tile

    // LDS byte base (addrspace(3) ptrtoint -> LDS offset)
    const unsigned sbase = (unsigned)(uintptr_t)(lds_bf16*)&smem[0];

    // staging pattern: row = tid/4 (64 rows per pass), seg = 16B quarter of 64B row
    const int srow = tid >> 2;
    const int sseg = tid & 3;

    auto stage = [&](int buf, int k0) {
        const unsigned abase = sbase + (unsigned)buf * BUF_BYTES;
        const unsigned bbase = abase + A_BYTES;
        // A tile: 256 rows x 64B
#pragma unroll
        for (int c = 0; c < 4; ++c) {
            const int r = srow + 64 * c;
            async_g2l_b128(abase + (unsigned)(r * (LDS_STRIDE * 2) + sseg * 16),
                           A + (size_t)(m0 + r) * K + k0 + sseg * 8);
        }
        // B tile: 64 rows x 64B
        async_g2l_b128(bbase + (unsigned)(srow * (LDS_STRIDE * 2) + sseg * 16),
                       B + (size_t)(n0 + srow) * K + k0 + sseg * 8);
    };

    v8f acc[2][4];
#pragma unroll
    for (int t = 0; t < 2; ++t)
#pragma unroll
        for (int j = 0; j < 4; ++j) acc[t][j] = v8f{};

    stage(0, 0);

    int buf = 0;
    for (int k0 = 0; k0 < K; k0 += BK, buf ^= 1) {
        wait_async0();          // my async copies for `buf` complete
        __syncthreads();        // everyone's copies complete; prev readers done
        if (k0 + BK < K) stage(buf ^ 1, k0 + BK);

        const __bf16* sA = &smem[buf * BUF_ELEMS];
        const __bf16* sB = &smem[buf * BUF_ELEMS + A_ELEMS];

        v16bf a0 = lds_a_frag(sA, mw, lane);
        v16bf a1 = lds_a_frag(sA, mw + 16, lane);
        v16bf bf[4];
#pragma unroll
        for (int j = 0; j < 4; ++j) bf[j] = lds_b_frag(sB, 16 * j, lane);

#pragma unroll
        for (int j = 0; j < 4; ++j)
            acc[0][j] = __builtin_amdgcn_wmma_f32_16x16x32_bf16(
                false, a0, false, bf[j], (short)0, acc[0][j], false, false);
#pragma unroll
        for (int j = 0; j < 4; ++j)
            acc[1][j] = __builtin_amdgcn_wmma_f32_16x16x32_bf16(
                false, a1, false, bf[j], (short)0, acc[1][j], false, false);
    }

    // C/D layout: VGPR r holds row (8*(lane>>4)+r), col = lane&15
    const int col = lane & 15;
#pragma unroll
    for (int t = 0; t < 2; ++t) {
        const int rowbase = m0 + mw + t * 16 + (lane >> 4) * 8;
#pragma unroll
        for (int j = 0; j < 4; ++j) {
            const int n = n0 + 16 * j + col;
#pragma unroll
            for (int r = 0; r < 8; ++r) {
                float v = acc[t][j][r];
                const size_t idx = (size_t)(rowbase + r) * N + n;
                if (MODE == 0) {
                    if (n < 2 * CC) {          // q and k columns get phi()
                        float z = v * TEMP_SCALE;
                        v = (z > 0.0f) ? (z + 1.0f) : expf(z);
                    }
                    Cf[idx] = v;
                } else if (MODE == 1) {
                    v += bias[n];
                    v = 0.5f * v * (1.0f + erff(v * 0.70710678118654752f));
                    Cb[idx] = (__bf16)v;
                } else {
                    v += bias[n];
                    Cf[idx] = v;
                }
            }
        }
    }
}

// ---------------------------------------------------------------------------
// Segment linear attention. One wave (32 lanes) per (n, h, segment).
// ---------------------------------------------------------------------------
__global__ void attn_kernel(const float* __restrict__ qkv,
                            const int* __restrict__ segoff,
                            float* __restrict__ t) {
    const int s    = blockIdx.x % SS;
    const int h    = (blockIdx.x / SS) % HH;
    const int n    = blockIdx.x / (SS * HH);
    const int lane = threadIdx.x;

    const int l0 = segoff[s];
    const int l1 = segoff[s + 1];
    const float* base = qkv + (size_t)n * LL * (3 * CC);
    const int co = h * DD;

    float kv[DD];
#pragma unroll
    for (int d = 0; d < DD; ++d) kv[d] = 0.0f;
    float ksum = 0.0f;

    for (int l = l0; l < l1; ++l) {
        const float* rowp = base + (size_t)l * (3 * CC);
        float kd = rowp[CC + co + lane];
        float vd = rowp[2 * CC + co + lane];
        ksum += kd;
#pragma unroll
        for (int d1 = 0; d1 < DD; ++d1)
            kv[d1] += __shfl(kd, d1, 32) * vd;
    }

    for (int l = l0; l < l1; ++l) {
        const float* rowp = base + (size_t)l * (3 * CC);
        float qd = rowp[co + lane];
        float z = qd * ksum;
#pragma unroll
        for (int off = 16; off; off >>= 1) z += __shfl_xor(z, off, 32);
        float num = 0.0f;
#pragma unroll
        for (int d1 = 0; d1 < DD; ++d1)
            num += __shfl(qd, d1, 32) * kv[d1];
        t[((size_t)n * LL + l) * CC + co + lane] = num / z;
    }
}

// ---------------------------------------------------------------------------
// beta gate + mix + power-norm 1. One block/token, 512 threads (wave = group).
// ---------------------------------------------------------------------------
__global__ void mix_pn1_kernel(const float* __restrict__ t,
                               const float* __restrict__ x,
                               const float* __restrict__ wb,
                               const float* __restrict__ pw,
                               const float* __restrict__ pb,
                               const float* __restrict__ pphi,
                               float* __restrict__ x1f,
                               __bf16* __restrict__ x1b) {
    const int tok = blockIdx.x;
    const int c   = threadIdx.x;
    const size_t idx = (size_t)tok * CC + c;

    const float tv = t[idx];
    const float xv = x[idx];

    float p = wb[c] * tv + wb[CC + c] * xv + wb[2 * CC + c] * (tv - xv);
#pragma unroll
    for (int off = 16; off; off >>= 1) p += __shfl_xor(p, off, 32);

    __shared__ float red[CC / 32];
    const int wave = c >> 5, lane = c & 31;
    if (lane == 0) red[wave] = p;
    __syncthreads();
    float dot = 0.0f;
#pragma unroll
    for (int i = 0; i < CC / 32; ++i) dot += red[i];
    const float beta = 1.0f / (1.0f + expf(-dot));

    const float y = beta * xv + (1.0f - beta) * tv;

    float q = y * y;
#pragma unroll
    for (int off = 16; off; off >>= 1) q += __shfl_xor(q, off, 32);
    const float m2 = q * (1.0f / 32.0f);

    const float gs = y * rsqrtf(m2 + EPS_PN);
    const float x1 = gs * rsqrtf(pphi[c] + EPS_PN) * pw[c] + pb[c];
    x1f[idx] = x1;
    x1b[idx] = (__bf16)x1;
}

// ---------------------------------------------------------------------------
// residual + power-norm 2 -> final output
// ---------------------------------------------------------------------------
__global__ void pn2_kernel(const float* __restrict__ ff,
                           const float* __restrict__ x1f,
                           const float* __restrict__ pw,
                           const float* __restrict__ pb,
                           const float* __restrict__ pphi,
                           float* __restrict__ out) {
    const int tok = blockIdx.x;
    const int c   = threadIdx.x;
    const size_t idx = (size_t)tok * CC + c;

    const float f = ff[idx] + x1f[idx];
    float q = f * f;
#pragma unroll
    for (int off = 16; off; off >>= 1) q += __shfl_xor(q, off, 32);
    const float m2 = q * (1.0f / 32.0f);
    const float gs = f * rsqrtf(m2 + EPS_PN);
    out[idx] = gs * rsqrtf(pphi[c] + EPS_PN) * pw[c] + pb[c];
}

// ---------------------------------------------------------------------------
// Host-side orchestration
// ---------------------------------------------------------------------------
extern "C" void kernel_launch(void* const* d_in, const int* in_sizes, int n_in,
                              void* d_out, int out_size, void* d_ws, size_t ws_size,
                              hipStream_t stream) {
    (void)in_sizes; (void)n_in; (void)out_size; (void)ws_size;

    const float* x       = (const float*)d_in[0];
    const int*   bi      = (const int*)  d_in[1];
    const float* w_qkv   = (const float*)d_in[2];
    const float* w_beta  = (const float*)d_in[3];
    const float* w1      = (const float*)d_in[4];
    const float* b1      = (const float*)d_in[5];
    const float* w2      = (const float*)d_in[6];
    const float* b2      = (const float*)d_in[7];
    const float* pn1_w   = (const float*)d_in[8];
    const float* pn1_b   = (const float*)d_in[9];
    const float* pn1_phi = (const float*)d_in[10];
    const float* pn2_w   = (const float*)d_in[11];
    const float* pn2_b   = (const float*)d_in[12];
    const float* pn2_phi = (const float*)d_in[13];
    float* out = (float*)d_out;

    unsigned char* ws = (unsigned char*)d_ws;
    __bf16* xb     = (__bf16*)(ws + 0);            // 4 MiB
    __bf16* wqkvb  = (__bf16*)(ws + (4u  << 20));  // 1.5 MiB
    __bf16* w1b    = (__bf16*)(ws + (6u  << 20));  // 0.5 MiB
    __bf16* w2b    = (__bf16*)(ws + ((6u << 20) + (512u << 10)));
    int*    segoff = (int*)   (ws + (7u  << 20));  // 17 ints
    float*  qkv    = (float*) (ws + (8u  << 20));  // 24 MiB (dead after attn)
    __bf16* hb     = (__bf16*)(ws + (8u  << 20));  // aliases qkv[0:4MiB]
    float*  ff     = (float*) (ws + (16u << 20));  // aliases qkv[8:16MiB]
    float*  tbuf   = (float*) (ws + (32u << 20));  // 8 MiB
    float*  x1f    = (float*) (ws + (40u << 20));  // 8 MiB
    __bf16* x1b    = (__bf16*)(ws + (48u << 20));  // 4 MiB

    // 1) fp32 -> bf16 conversions
    {
        int n;
        n = TOKENS * CC;
        f32_to_bf16_kernel<<<(n + 255) / 256, 256, 0, stream>>>(x, xb, n);
        n = 3 * CC * CC;
        f32_to_bf16_kernel<<<(n + 255) / 256, 256, 0, stream>>>(w_qkv, wqkvb, n);
        n = CC * CC;
        f32_to_bf16_kernel<<<(n + 255) / 256, 256, 0, stream>>>(w1, w1b, n);
        f32_to_bf16_kernel<<<(n + 255) / 256, 256, 0, stream>>>(w2, w2b, n);
    }

    // 2) segment boundaries
    seg_bounds_kernel<<<1, 32, 0, stream>>>(bi, segoff);

    // 3) QKV GEMM (4096x512 * 512x1536) with fused phi on q,k
    {
        dim3 grid(TOKENS / BM, (3 * CC) / BN);
        gemm_bf16_wmma_kernel<0><<<grid, 256, 0, stream>>>(
            xb, wqkvb, nullptr, qkv, nullptr, TOKENS, 3 * CC, CC);
    }

    // 4) segment linear attention -> t
    attn_kernel<<<NN * HH * SS, 32, 0, stream>>>(qkv, segoff, tbuf);

    // 5) beta gate + mix + power-norm 1 -> x1 (fp32 + bf16)
    mix_pn1_kernel<<<TOKENS, CC, 0, stream>>>(
        tbuf, x, w_beta, pn1_w, pn1_b, pn1_phi, x1f, x1b);

    // 6) FFN GEMM 1: gelu(x1 @ w1^T + b1) -> bf16
    {
        dim3 grid(TOKENS / BM, CC / BN);
        gemm_bf16_wmma_kernel<1><<<grid, 256, 0, stream>>>(
            x1b, w1b, b1, nullptr, hb, TOKENS, CC, CC);
    }

    // 7) FFN GEMM 2: h @ w2^T + b2 -> fp32
    {
        dim3 grid(TOKENS / BM, CC / BN);
        gemm_bf16_wmma_kernel<2><<<grid, 256, 0, stream>>>(
            hb, w2b, b2, ff, nullptr, TOKENS, CC, CC);
    }

    // 8) residual + power-norm 2 -> output
    pn2_kernel<<<TOKENS, CC, 0, stream>>>(ff, x1f, pn2_w, pn2_b, pn2_phi, out);
}
